// TransformerBlock_1906965479563
// MI455X (gfx1250) — compile-verified, matrix operand order checked by reading, not run on hardware
//
#include <hip/hip_runtime.h>
#include <cstdint>

// ---- fixed problem dims (from reference) ----
// B=8, S=1024, H=1024, NH=16, D=64, FF=4096, M = B*S = 8192

typedef __bf16 bf;
typedef __attribute__((ext_vector_type(8)))  __bf16 v8bf;
typedef __attribute__((ext_vector_type(16))) __bf16 v16bf;
typedef __attribute__((ext_vector_type(8)))  float  v8f;

#define USE_ASYNC_LDS 1   // global_load_async_to_lds_b128 pipeline (verified: assembles)

__device__ __forceinline__ v16bf ld16(const bf* p0, const bf* p1) {
    v8bf a = *(const v8bf*)p0;
    v8bf b = *(const v8bf*)p1;
    v16bf r;
#pragma unroll
    for (int i = 0; i < 8; ++i) { r[i] = a[i]; r[i + 8] = b[i]; }
    return r;
}

__device__ __forceinline__ v8f wmma_bf16(v16bf a, v16bf b, v8f c) {
    // D = A(16x32 bf16) * B(32x16 bf16) + C(16x16 f32)
    return __builtin_amdgcn_wmma_f32_16x16x32_bf16(false, a, false, b, (short)0, c,
                                                   false, false);
}

// low 32 bits of a generic pointer into __shared__ == LDS byte offset (ISA 10.2)
__device__ __forceinline__ uint32_t lds_lo32(const void* p) {
    return (uint32_t)(uintptr_t)p;
}

__device__ __forceinline__ void async_b128(const void* gptr, uint32_t ldsaddr) {
#if USE_ASYNC_LDS
    asm volatile("global_load_async_to_lds_b128 %0, %1, off"
                 :: "v"(ldsaddr), "v"(gptr) : "memory");
#endif
}

__device__ __forceinline__ void wait_async_le0() {
#if USE_ASYNC_LDS
#if __has_builtin(__builtin_amdgcn_s_wait_asynccnt)
    __builtin_amdgcn_s_wait_asynccnt(0);
#else
    asm volatile("s_wait_asynccnt 0x0" ::: "memory");
#endif
#endif
}
__device__ __forceinline__ void wait_async_le2() {
#if USE_ASYNC_LDS
#if __has_builtin(__builtin_amdgcn_s_wait_asynccnt)
    __builtin_amdgcn_s_wait_asynccnt(2);
#else
    asm volatile("s_wait_asynccnt 0x2" ::: "memory");
#endif
#endif
}
__device__ __forceinline__ void wait_async_le3() {
#if USE_ASYNC_LDS
#if __has_builtin(__builtin_amdgcn_s_wait_asynccnt)
    __builtin_amdgcn_s_wait_asynccnt(3);
#else
    asm volatile("s_wait_asynccnt 0x3" ::: "memory");
#endif
#endif
}

// ---------------------------------------------------------------------------
// Weight convert + transpose: w[K][N] f32  ->  wt[N][K] bf16
// ---------------------------------------------------------------------------
template <int K, int N>
__global__ __launch_bounds__(256) void cvt_wt(const float* __restrict__ w,
                                              bf* __restrict__ wt) {
    int i = blockIdx.x * 256 + threadIdx.x;   // i < K*N (K*N divisible by 256)
    int k = i / N;
    int n = i - k * N;
    wt[(size_t)n * K + k] = (bf)w[i];
}

// ---------------------------------------------------------------------------
// LayerNorm over H=1024, f32 in -> bf16 out. One row per block, 256 threads.
// ---------------------------------------------------------------------------
__global__ __launch_bounds__(256) void ln_bf16(const float* __restrict__ x,
                                               const float* __restrict__ g,
                                               const float* __restrict__ b,
                                               bf* __restrict__ out) {
    __shared__ float red[256];
    const int row = blockIdx.x;
    const int t = threadIdx.x;
    const float* xr = x + (size_t)row * 1024;
    float4 v = ((const float4*)xr)[t];

    red[t] = v.x + v.y + v.z + v.w;
    __syncthreads();
#pragma unroll
    for (int o = 128; o > 0; o >>= 1) {
        if (t < o) red[t] += red[t + o];
        __syncthreads();
    }
    float mean = red[0] * (1.0f / 1024.0f);
    __syncthreads();

    float dx = v.x - mean, dy = v.y - mean, dz = v.z - mean, dw = v.w - mean;
    red[t] = dx * dx + dy * dy + dz * dz + dw * dw;
    __syncthreads();
#pragma unroll
    for (int o = 128; o > 0; o >>= 1) {
        if (t < o) red[t] += red[t + o];
        __syncthreads();
    }
    float rstd = rsqrtf(red[0] * (1.0f / 1024.0f) + 1e-5f);

    bf* orow = out + (size_t)row * 1024;
    int i = t * 4;
    orow[i + 0] = (bf)(dx * rstd * g[i + 0] + b[i + 0]);
    orow[i + 1] = (bf)(dy * rstd * g[i + 1] + b[i + 1]);
    orow[i + 2] = (bf)(dz * rstd * g[i + 2] + b[i + 2]);
    orow[i + 3] = (bf)(dw * rstd * g[i + 3] + b[i + 3]);
}

// ---------------------------------------------------------------------------
// Tiled bf16 WMMA GEMM:  C[M][N] = (A[M][K] * BT[N][K]^T + bias) * oscale
// Block tile 128(M) x 64(N), 8 waves as 4(M) x 2(N), each wave 32x32 (4 accs).
// K staged through double-buffered LDS (async global->LDS, ASYNCcnt).
// LDS rows padded to 40 bf16 (20 banks) -> conflict-free fragment reads.
// MODE 0: f32 out (+ optional residual)      MODE 1: bf16 row-major out
// MODE 2: bf16 out stored as V^T (B,NH,D,S)
// grid = (N/64, M/128), block = 256.
// ---------------------------------------------------------------------------
#define LDSP 40   // padded row stride (elements) for 32-wide K slice

template <int M, int N, int K, int MODE, bool GELU, bool RES>
__global__ __launch_bounds__(256) void gemm_bf16(const bf* __restrict__ A,
                                                 const bf* __restrict__ BT,
                                                 const float* __restrict__ bias,
                                                 const float* __restrict__ res,
                                                 float* __restrict__ outf,
                                                 bf* __restrict__ outb,
                                                 float oscale) {
    __shared__ bf As[2][128 * LDSP];
    __shared__ bf Bs[2][64 * LDSP];

    const int tid = threadIdx.x;
    const int lane = tid & 31;
    const int wave = tid >> 5;
    const int wm = wave & 3;        // 0..3  (M direction)
    const int wn = wave >> 2;       // 0..1  (N direction)
    const int tM = blockIdx.y * 128;
    const int tN = blockIdx.x * 64;
    const int rc = lane & 15;       // row/col within 16-tile
    const int hl = lane >> 4;       // K-half select
    const int k0 = hl * 8;

    // staging map: 16B chunks; A tile 128x32 (2 chunks/thread), B tile 64x32 (1)
    const int arow0 = (tid) >> 2, ach0 = tid & 3;
    const int arow1 = (tid + 256) >> 2, ach1 = tid & 3;
    const int brow = tid >> 2, bch = tid & 3;

    auto stage = [&](int buf, int kb) {
        const bf* ga0 = A + (size_t)(tM + arow0) * K + kb + ach0 * 8;
        const bf* ga1 = A + (size_t)(tM + arow1) * K + kb + ach1 * 8;
        const bf* gb = BT + (size_t)(tN + brow) * K + kb + bch * 8;
        bf* la0 = &As[buf][arow0 * LDSP + ach0 * 8];
        bf* la1 = &As[buf][arow1 * LDSP + ach1 * 8];
        bf* lb = &Bs[buf][brow * LDSP + bch * 8];
#if USE_ASYNC_LDS
        async_b128(ga0, lds_lo32(la0));
        async_b128(ga1, lds_lo32(la1));
        async_b128(gb, lds_lo32(lb));
#else
        *(v8bf*)la0 = *(const v8bf*)ga0;
        *(v8bf*)la1 = *(const v8bf*)ga1;
        *(v8bf*)lb = *(const v8bf*)gb;
#endif
    };

    v8f acc00 = {}, acc01 = {}, acc10 = {}, acc11 = {};

    stage(0, 0);
    for (int kb = 0; kb < K; kb += 32) {
        const int cur = (kb >> 5) & 1;
        const bool more = (kb + 32) < K;
        if (more) stage(cur ^ 1, kb + 32);
#if USE_ASYNC_LDS
        if (more) wait_async_le3(); else wait_async_le0();
#endif
        __syncthreads();

        const bf* a0p = &As[cur][(wm * 32 + rc) * LDSP];
        const bf* a1p = &As[cur][(wm * 32 + 16 + rc) * LDSP];
        const bf* b0p = &Bs[cur][(wn * 32 + rc) * LDSP];
        const bf* b1p = &Bs[cur][(wn * 32 + 16 + rc) * LDSP];
        v16bf a0 = ld16(a0p + k0, a0p + 16 + k0);
        v16bf a1 = ld16(a1p + k0, a1p + 16 + k0);
        v16bf b0 = ld16(b0p + k0, b0p + 16 + k0);
        v16bf b1 = ld16(b1p + k0, b1p + 16 + k0);
        acc00 = wmma_bf16(a0, b0, acc00);
        acc01 = wmma_bf16(a0, b1, acc01);
        acc10 = wmma_bf16(a1, b0, acc10);
        acc11 = wmma_bf16(a1, b1, acc11);
        __syncthreads();
    }

    const float bv0 = bias[tN + wn * 32 + rc];
    const float bv1 = bias[tN + wn * 32 + 16 + rc];

    auto emit = [&](v8f acc, int mo, int no, float bv) {
#pragma unroll
        for (int r = 0; r < 8; ++r) {
            float v = (acc[r] + bv) * oscale;
            if (GELU) v = 0.5f * v * (1.0f + erff(v * 0.70710678118654752f));
            const int mi = tM + wm * 32 + mo + r + 8 * hl;
            const int ni = tN + wn * 32 + no + rc;
            if (MODE == 0) {
                float o = v;
                if (RES) o += res[(size_t)mi * N + ni];
                outf[(size_t)mi * N + ni] = o;
            } else if (MODE == 1) {
                outb[(size_t)mi * N + ni] = (bf)v;
            } else {  // MODE 2: V^T layout (B, NH, D, S)
                const int bb = mi >> 10, ss = mi & 1023;   // S = 1024
                const int nh = ni >> 6, dd = ni & 63;      // D = 64
                outb[((size_t)((bb * 16 + nh) * 64 + dd)) * 1024 + ss] = (bf)v;
            }
        }
    };
    emit(acc00, 0, 0, bv0);
    emit(acc01, 0, 16, bv1);
    emit(acc10, 16, 0, bv0);
    emit(acc11, 16, 16, bv1);
}

// ---------------------------------------------------------------------------
// Flash attention: q,k in (B,S,H)=(B,S,NH,D) bf16 (q pre-scaled by 1/sqrt(D)),
// vT in (B,NH,D,S) bf16.  All 8 waves of a block share one (b,nh) and the same
// K/V key-step tiles, so K (32x64) and V^T (64x32) are staged per step into
// double-buffered LDS via async global->LDS (2 async b128 per thread/stage).
// Each wave owns one 16-query tile. grid = (S/128, B*NH), block = 256.
// K rows padded to 72 elems, V rows to 40 -> conflict-free fragment reads.
// ---------------------------------------------------------------------------
__global__ __launch_bounds__(256) void attn_kernel(const bf* __restrict__ q,
                                                   const bf* __restrict__ kk,
                                                   const bf* __restrict__ vT,
                                                   const unsigned char* __restrict__ mask,
                                                   bf* __restrict__ ctx) {
    __shared__ bf Ks[2][32 * 72];
    __shared__ bf Vs[2][64 * 40];
    __shared__ bf plds[8][16][32];

    const int tid = threadIdx.x;
    const int lane = tid & 31;
    const int wave = tid >> 5;
    const int rc = lane & 15;
    const int hl = lane >> 4;
    const int k0 = hl * 8;

    const int bh = blockIdx.y;
    const int b = bh >> 4;          // NH = 16
    const int nh = bh & 15;
    const int q0 = (blockIdx.x * 8 + wave) * 16;

    // Q A-fragments (row = q0+rc, head-dim as K; D=64 -> two K=32 steps)
    const bf* qp = q + ((size_t)(b * 1024 + q0 + rc) * 1024 + nh * 64);
    const v16bf qa0 = ld16(qp + k0, qp + 16 + k0);
    const v16bf qa1 = ld16(qp + 32 + k0, qp + 48 + k0);

    const bf* kbase = kk + (size_t)b * 1024 * 1024 + nh * 64;
    const bf* vbase = vT + (size_t)(b * 16 + nh) * 64 * 1024;
    const unsigned char* mrow = mask + b * 1024;

    // staging maps (16B chunks): K tile 32 rows x 8 chunks; V tile 64 rows x 4
    const int kj = tid >> 3, kch = tid & 7;
    const int vd = tid >> 2, vch = tid & 3;

    auto stage = [&](int buf, int jb) {
        const bf* gk = kbase + (size_t)(jb + kj) * 1024 + kch * 8;
        const bf* gv = vbase + (size_t)vd * 1024 + jb + vch * 8;
        bf* lk = &Ks[buf][kj * 72 + kch * 8];
        bf* lv = &Vs[buf][vd * 40 + vch * 8];
#if USE_ASYNC_LDS
        async_b128(gk, lds_lo32(lk));
        async_b128(gv, lds_lo32(lv));
#else
        *(v8bf*)lk = *(const v8bf*)gk;
        *(v8bf*)lv = *(const v8bf*)gv;
#endif
    };

    float mi[8], li[8];
    v8f o0 = {}, o1 = {}, o2 = {}, o3 = {};
#pragma unroll
    for (int r = 0; r < 8; ++r) { mi[r] = -1e30f; li[r] = 0.0f; }

    stage(0, 0);
    for (int jb = 0; jb < 1024; jb += 32) {
        const int cur = (jb >> 5) & 1;
        const bool more = (jb + 32) < 1024;
        if (more) stage(cur ^ 1, jb + 32);
#if USE_ASYNC_LDS
        if (more) wait_async_le2(); else wait_async_le0();
#endif
        __syncthreads();

        // --- scores: two 16x16 tiles (keys jb..jb+15, jb+16..jb+31) ---
        v8f s0 = {}, s1 = {};
        const bf* kc0 = &Ks[cur][rc * 72];
        s0 = wmma_bf16(qa0, ld16(kc0 + k0, kc0 + 16 + k0), s0);
        s0 = wmma_bf16(qa1, ld16(kc0 + 32 + k0, kc0 + 48 + k0), s0);
        const bf* kc1 = &Ks[cur][(16 + rc) * 72];
        s1 = wmma_bf16(qa0, ld16(kc1 + k0, kc1 + 16 + k0), s1);
        s1 = wmma_bf16(qa1, ld16(kc1 + 32 + k0, kc1 + 48 + k0), s1);

        const float madd0 = mrow[jb + rc] ? 0.0f : -1e30f;
        const float madd1 = mrow[jb + 16 + rc] ? 0.0f : -1e30f;

        // --- online softmax (rows r+8*hl; reduce over 16 lanes per half) ---
        // (q was pre-scaled by 1/sqrt(D) in its projection epilogue)
#pragma unroll
        for (int r = 0; r < 8; ++r) {
            float a0 = s0[r] + madd0;
            float a1 = s1[r] + madd1;
            float mx = fmaxf(a0, a1);
            mx = fmaxf(mx, __shfl_xor(mx, 1));
            mx = fmaxf(mx, __shfl_xor(mx, 2));
            mx = fmaxf(mx, __shfl_xor(mx, 4));
            mx = fmaxf(mx, __shfl_xor(mx, 8));
            const float mnew = fmaxf(mi[r], mx);
            const float alpha = __expf(mi[r] - mnew);
            const float p0 = __expf(a0 - mnew);
            const float p1 = __expf(a1 - mnew);
            float rs = p0 + p1;
            rs += __shfl_xor(rs, 1);
            rs += __shfl_xor(rs, 2);
            rs += __shfl_xor(rs, 4);
            rs += __shfl_xor(rs, 8);
            li[r] = li[r] * alpha + rs;
            mi[r] = mnew;
            o0[r] *= alpha; o1[r] *= alpha; o2[r] *= alpha; o3[r] *= alpha;
            s0[r] = p0; s1[r] = p1;
        }

        // --- C-fragment -> A-fragment for P via per-wave LDS region ---
        // (same-wave RAW: hardware-ordered via DScnt waits inserted by compiler)
#pragma unroll
        for (int r = 0; r < 8; ++r) {
            plds[wave][r + 8 * hl][rc] = (bf)s0[r];
            plds[wave][r + 8 * hl][16 + rc] = (bf)s1[r];
        }
        const v16bf pa = ld16(&plds[wave][rc][k0], &plds[wave][rc][16 + k0]);

        // --- P(16x32) * V(32x64): 4 N-chunks of 16 ---
        const bf* v0p = &Vs[cur][(0 * 16 + rc) * 40];
        o0 = wmma_bf16(pa, ld16(v0p + k0, v0p + 16 + k0), o0);
        const bf* v1p = &Vs[cur][(1 * 16 + rc) * 40];
        o1 = wmma_bf16(pa, ld16(v1p + k0, v1p + 16 + k0), o1);
        const bf* v2p = &Vs[cur][(2 * 16 + rc) * 40];
        o2 = wmma_bf16(pa, ld16(v2p + k0, v2p + 16 + k0), o2);
        const bf* v3p = &Vs[cur][(3 * 16 + rc) * 40];
        o3 = wmma_bf16(pa, ld16(v3p + k0, v3p + 16 + k0), o3);

        __syncthreads();   // all reads of Ks/Vs[cur] + plds done before restage
    }

    // --- normalize + store ctx (B,S,H) bf16 ---
#pragma unroll
    for (int r = 0; r < 8; ++r) {
        const int m = q0 + r + 8 * hl;
        const float inv = 1.0f / li[r];
        bf* crow = ctx + ((size_t)(b * 1024 + m) * 1024 + nh * 64);
        crow[0 * 16 + rc] = (bf)(o0[r] * inv);
        crow[1 * 16 + rc] = (bf)(o1[r] * inv);
        crow[2 * 16 + rc] = (bf)(o2[r] * inv);
        crow[3 * 16 + rc] = (bf)(o3[r] * inv);
    }
}

// ---------------------------------------------------------------------------
// Host orchestration
// ---------------------------------------------------------------------------
extern "C" void kernel_launch(void* const* d_in, const int* in_sizes, int n_in,
                              void* d_out, int out_size, void* d_ws, size_t ws_size,
                              hipStream_t stream) {
    const float* x   = (const float*)d_in[0];
    const unsigned char* mask = (const unsigned char*)d_in[1];
    const float* ln1g = (const float*)d_in[2];
    const float* ln1b = (const float*)d_in[3];
    const float* wq = (const float*)d_in[4];
    const float* bq = (const float*)d_in[5];
    const float* wk = (const float*)d_in[6];
    const float* bk = (const float*)d_in[7];
    const float* wv = (const float*)d_in[8];
    const float* bv = (const float*)d_in[9];
    const float* wo = (const float*)d_in[10];
    const float* bo = (const float*)d_in[11];
    const float* ln2g = (const float*)d_in[12];
    const float* ln2b = (const float*)d_in[13];
    const float* w1 = (const float*)d_in[14];
    const float* b1 = (const float*)d_in[15];
    const float* w2 = (const float*)d_in[16];
    const float* b2 = (const float*)d_in[17];

    char* ws = (char*)d_ws;
    size_t off = 0;
    auto alloc = [&](size_t bytes) -> void* {
        void* p = ws + off;
        off += (bytes + 255) & ~(size_t)255;
        return p;
    };

    const size_t MM = 8192;  // B*S
    bf* wqT = (bf*)alloc(1024ull * 1024 * sizeof(bf));
    bf* wkT = (bf*)alloc(1024ull * 1024 * sizeof(bf));
    bf* wvT = (bf*)alloc(1024ull * 1024 * sizeof(bf));
    bf* woT = (bf*)alloc(1024ull * 1024 * sizeof(bf));
    bf* w1T = (bf*)alloc(1024ull * 4096 * sizeof(bf));
    bf* w2T = (bf*)alloc(4096ull * 1024 * sizeof(bf));
    bf* hbf = (bf*)alloc(MM * 1024 * sizeof(bf));
    bf* qbuf = (bf*)alloc(MM * 1024 * sizeof(bf));
    bf* kbuf = (bf*)alloc(MM * 1024 * sizeof(bf));
    bf* vtbuf = (bf*)alloc(MM * 1024 * sizeof(bf));
    bf* ctxb = (bf*)alloc(MM * 1024 * sizeof(bf));
    float* x1 = (float*)alloc(MM * 1024 * sizeof(float));
    bf* h2b = (bf*)alloc(MM * 1024 * sizeof(bf));
    bf* ff1b = (bf*)alloc(MM * 4096 * sizeof(bf));

    // weights -> bf16 transposed
    cvt_wt<1024, 1024><<<4096, 256, 0, stream>>>(wq, wqT);
    cvt_wt<1024, 1024><<<4096, 256, 0, stream>>>(wk, wkT);
    cvt_wt<1024, 1024><<<4096, 256, 0, stream>>>(wv, wvT);
    cvt_wt<1024, 1024><<<4096, 256, 0, stream>>>(wo, woT);
    cvt_wt<1024, 4096><<<16384, 256, 0, stream>>>(w1, w1T);
    cvt_wt<4096, 1024><<<16384, 256, 0, stream>>>(w2, w2T);

    // LN1
    ln_bf16<<<8192, 256, 0, stream>>>(x, ln1g, ln1b, hbf);

    // QKV projections (grid = N/64, M/128); q pre-scaled by 1/sqrt(D)=0.125
    gemm_bf16<8192, 1024, 1024, 1, false, false>
        <<<dim3(16, 64), 256, 0, stream>>>(hbf, wqT, bq, nullptr, nullptr, qbuf, 0.125f);
    gemm_bf16<8192, 1024, 1024, 1, false, false>
        <<<dim3(16, 64), 256, 0, stream>>>(hbf, wkT, bk, nullptr, nullptr, kbuf, 1.0f);
    gemm_bf16<8192, 1024, 1024, 2, false, false>
        <<<dim3(16, 64), 256, 0, stream>>>(hbf, wvT, bv, nullptr, nullptr, vtbuf, 1.0f);

    // attention
    attn_kernel<<<dim3(8, 128), 256, 0, stream>>>(qbuf, kbuf, vtbuf, mask, ctxb);

    // output projection + residual -> x1
    gemm_bf16<8192, 1024, 1024, 0, false, true>
        <<<dim3(16, 64), 256, 0, stream>>>(ctxb, woT, bo, x, x1, nullptr, 1.0f);

    // LN2
    ln_bf16<<<8192, 256, 0, stream>>>(x1, ln2g, ln2b, h2b);

    // FFN
    gemm_bf16<8192, 4096, 1024, 1, true, false>
        <<<dim3(64, 64), 256, 0, stream>>>(h2b, w1T, b1, nullptr, nullptr, ff1b, 1.0f);
    gemm_bf16<8192, 1024, 4096, 0, false, true>
        <<<dim3(16, 64), 256, 0, stream>>>(ff1b, w2T, b2, x1, (float*)d_out, nullptr, 1.0f);
}
